// LinearGate_1108101562616
// MI455X (gfx1250) — matrix operand order repaired and checked by reading, compiled, not run on hardware
//
#include <hip/hip_runtime.h>
#include <hip/hip_bf16.h>
#include <math.h>

#define N_TOKENS  16384
#define D_MODEL   4096
#define N_EXPERTS 64
#define TOP_K     8

#define KSLICES         4
#define KS_LEN          (D_MODEL / KSLICES)   // 1024 per wave
#define TILES_PER_BLOCK 4
#define WAVES_PER_BLOCK (TILES_PER_BLOCK * KSLICES)   // 16 waves = 512 threads

typedef __attribute__((ext_vector_type(16))) __bf16 v16bf;
typedef __attribute__((ext_vector_type(8)))  __bf16 v8bf;
typedef __attribute__((ext_vector_type(8)))  float  v8f;

// fp32 -> bf16 RNE for the W pre-pass (bit-exact, scalar path)
static __device__ __forceinline__ unsigned short f2bf_bits(float f) {
    unsigned u = __builtin_bit_cast(unsigned, f);
    unsigned r = u + 0x7FFFu + ((u >> 16) & 1u);
    return (unsigned short)(r >> 16);
}

// Packed f32->bf16 for the hot loop (lowered to v_cvt_pk_bf16_f32).
static __device__ __forceinline__ void cvt2(v16bf& a, int i, float lo, float hi) {
#if __has_builtin(__builtin_amdgcn_cvt_pk_bf16_f32)
    auto p = __builtin_amdgcn_cvt_pk_bf16_f32(lo, hi);
    a[2 * i]     = p[0];
    a[2 * i + 1] = p[1];
#else
    a[2 * i]     = (__bf16)lo;
    a[2 * i + 1] = (__bf16)hi;
#endif
}

// ---------------------------------------------------------------------------
// Pre-pass: convert W [64 x 4096] fp32 -> bf16 in workspace (512 KB, L2-hot)
// ---------------------------------------------------------------------------
__global__ void convert_w_bf16(const float* __restrict__ W,
                               unsigned short* __restrict__ Wb, int n) {
    int i = blockIdx.x * blockDim.x + threadIdx.x;
    if (i < n) Wb[i] = f2bf_bits(W[i]);
}

// One k-step (K=32) worth of operands, kept entirely in registers.
struct Frag {
    float4 f0, f1, f2, f3;   // A: 16 fp32 for this lane
    v8bf   b[4][2];          // B: 4 N-tiles x 16 bf16 for this lane

    __device__ __forceinline__ void load(const float* __restrict__ ap,
                                         const __bf16* __restrict__ bp) {
        f0 = *reinterpret_cast<const float4*>(ap);
        f1 = *reinterpret_cast<const float4*>(ap + 4);
        f2 = *reinterpret_cast<const float4*>(ap + 16);
        f3 = *reinterpret_cast<const float4*>(ap + 20);
#pragma unroll
        for (int t = 0; t < 4; ++t) {
            const __bf16* p = bp + (size_t)t * 16 * D_MODEL;
            b[t][0] = *reinterpret_cast<const v8bf*>(p);
            b[t][1] = *reinterpret_cast<const v8bf*>(p + 16);
        }
    }

    __device__ __forceinline__ void compute(v8f acc[4]) {
        float af[16] = {f0.x, f0.y, f0.z, f0.w, f1.x, f1.y, f1.z, f1.w,
                        f2.x, f2.y, f2.z, f2.w, f3.x, f3.y, f3.z, f3.w};
        v16bf a;
#pragma unroll
        for (int i = 0; i < 8; ++i) cvt2(a, i, af[2 * i], af[2 * i + 1]);

#pragma unroll
        for (int t = 0; t < 4; ++t) {
            v16bf bt = __builtin_shufflevector(b[t][0], b[t][1],
                                               0, 1, 2, 3, 4, 5, 6, 7,
                                               8, 9, 10, 11, 12, 13, 14, 15);
            acc[t] = __builtin_amdgcn_wmma_f32_16x16x32_bf16(
                false, a, false, bt, (short)0, acc[t], false, false);
        }
    }
};

// ---------------------------------------------------------------------------
// Main kernel. Wave w: token-tile = w>>2, K-slice = w&3 (K-split x4 for
// occupancy / HBM latency hiding). Each wave: 16 tokens x 64 experts partial
// GEMM via v_wmma_f32_16x16x32_bf16, double-buffered streams + WGP-scope
// prefetch. Slice-0 wave reduces 4 partials in LDS and runs top-8.
// ---------------------------------------------------------------------------
__global__ __launch_bounds__(WAVES_PER_BLOCK * 32)
void moe_gate_topk(const float* __restrict__ x,
                   const unsigned short* __restrict__ Wb_u16,
                   int* __restrict__ out) {
    const __bf16* __restrict__ Wb = reinterpret_cast<const __bf16*>(Wb_u16);

    __shared__ float lds_part[WAVES_PER_BLOCK][16][N_EXPERTS];   // 64 KB

    const int tid    = threadIdx.x;
    const int wave   = tid >> 5;
    const int lane   = tid & 31;
    const int half   = lane >> 4;     // 0: lanes 0-15, 1: lanes 16-31
    const int l16    = lane & 15;
    const int tile   = wave >> 2;     // 0..3 -> which 16 tokens
    const int kslice = wave & 3;      // 0..3 -> which quarter of K

    const int token_base = (blockIdx.x * TILES_PER_BLOCK + tile) * 16;
    const int kbeg       = kslice * KS_LEN;

    // A (16x32 bf16 ISA layout): lane row = l16, K chunks at half*8 / +16.
    const float* __restrict__ aptr =
        x + (size_t)(token_base + l16) * D_MODEL + kbeg + half * 8;
    // B (32x16 bf16 ISA layout): lane column = l16, K chunks at half*8 / +16.
    const __bf16* __restrict__ bptr =
        Wb + (size_t)l16 * D_MODEL + kbeg + half * 8;

    v8f acc[4] = {v8f{0.f}, v8f{0.f}, v8f{0.f}, v8f{0.f}};

    Frag g0, g1;
    g0.load(aptr, bptr);                       // prologue: k = kbeg

    // Two k-steps per iteration so each buffer index is compile-time constant.
    for (int k0 = 0; k0 < KS_LEN; k0 += 64) {
        g1.load(aptr + k0 + 32, bptr + k0 + 32);

        // WGP-scope prefetch (non-speculative, no LOADcnt): two cachelines
        // ~2 iterations ahead covers the 256 B/row/iteration A stream.
        if (k0 + 192 < KS_LEN) {
            const float* pf = aptr + k0 + 128;
            asm volatile("global_prefetch_b8 %0, off" :: "v"(pf));
            asm volatile("global_prefetch_b8 %0, off offset:128" :: "v"(pf));
        }

        g0.compute(acc);                               // consume k0
        if (k0 + 64 < KS_LEN)
            g0.load(aptr + k0 + 64, bptr + k0 + 64);   // next even step
        g1.compute(acc);                               // consume k0+32
    }

    // ---- spill partial logits to LDS per the C/D layout:
    // VGPR r: lanes 0-15 -> (M=r, N=lane), lanes 16-31 -> (M=8+r, N=lane-16)
#pragma unroll
    for (int t = 0; t < 4; ++t)
#pragma unroll
        for (int r = 0; r < 8; ++r)
            lds_part[wave][half * 8 + r][t * 16 + l16] = acc[t][r];

    __syncthreads();

    // ---- K-slice reduction + top-8 (argmax over logits == over softmax).
    // Strict '>' matches jax.lax.top_k tie-break (lowest index wins).
    if (kslice == 0 && lane < 16) {
        const int m = lane;
        float4* r0 = reinterpret_cast<float4*>(&lds_part[tile * 4 + 0][m][0]);
        float4* r1 = reinterpret_cast<float4*>(&lds_part[tile * 4 + 1][m][0]);
        float4* r2 = reinterpret_cast<float4*>(&lds_part[tile * 4 + 2][m][0]);
        float4* r3 = reinterpret_cast<float4*>(&lds_part[tile * 4 + 3][m][0]);
#pragma unroll
        for (int e = 0; e < N_EXPERTS / 4; ++e) {
            float4 a = r0[e], b = r1[e], c = r2[e], d = r3[e];
            a.x += b.x + c.x + d.x;
            a.y += b.y + c.y + d.y;
            a.z += b.z + c.z + d.z;
            a.w += b.w + c.w + d.w;
            r0[e] = a;
        }

        float* row = &lds_part[tile * 4][m][0];
        const int obase = (token_base + m) * TOP_K;
#pragma unroll 1
        for (int k = 0; k < TOP_K; ++k) {
            float best = row[0];
            int   bi   = 0;
#pragma unroll 1
            for (int e = 1; e < N_EXPERTS; ++e) {
                float v = row[e];
                if (v > best) { best = v; bi = e; }
            }
            out[obase + k] = bi;
            row[bi] = -INFINITY;
        }
    }
}

// ---------------------------------------------------------------------------
extern "C" void kernel_launch(void* const* d_in, const int* in_sizes, int n_in,
                              void* d_out, int out_size, void* d_ws,
                              size_t ws_size, hipStream_t stream) {
    const float* x = (const float*)d_in[0];   // [16384, 4096] fp32
    const float* W = (const float*)d_in[1];   // [64, 4096] fp32
    int* out = (int*)d_out;                   // [16384, 8] int32
    unsigned short* Wb = (unsigned short*)d_ws;

    const int nW = N_EXPERTS * D_MODEL;
    convert_w_bf16<<<(nW + 255) / 256, 256, 0, stream>>>(W, Wb, nW);

    const int tokens_per_block = TILES_PER_BLOCK * 16;   // 64
    moe_gate_topk<<<N_TOKENS / tokens_per_block, WAVES_PER_BLOCK * 32, 0,
                    stream>>>(x, Wb, out);
}
